// LightweightRGTAN_43817256354327
// MI455X (gfx1250) — matrix operand order repaired and checked by reading, not compiled, not run on hardware
//
#include <hip/hip_runtime.h>
#include <hip/hip_bf16.h>

#define NN 100000
#define EE 1000000
#define HID 64

// ROCm 7.2 (clang-22) exposes the 5-arg tensor_load_to_lds builtin and does NOT
// ship the TDM header; TheRock toolchain ships the header (6-arg form).
#if !__has_include(<hip/amd_detail/amd_gfx1250_TDM.h>)
#define USE_TDM_BUILTIN 1
#else
#define USE_TDM_BUILTIN 0
#endif

typedef __attribute__((ext_vector_type(16))) _Float16 v16h;
typedef __attribute__((ext_vector_type(8)))  _Float16 v8h;
typedef __attribute__((ext_vector_type(8)))  float    v8f;
typedef __attribute__((ext_vector_type(4)))  unsigned u32x4;
typedef __attribute__((ext_vector_type(8)))  int      i32x8;
typedef __attribute__((ext_vector_type(4)))  int      i32x4;

// ---------------------------------------------------------------------------
// WMMA GEMM: Y[N x 64] = X[N x K] @ W[K x 64] + bias, optional ReLU.
// Block = 128 threads (4 waves). Block owns a 16-row tile; wave w owns the
// 16-column tile w. X tile and transposed weights staged in LDS (f16);
// K<=128 weight tiles are brought in via the Tensor Data Mover.
// K is compile-time -> fully unrolled WMMA chain (K/32 v_wmma per wave).
// ---------------------------------------------------------------------------
template<int K>
__global__ __launch_bounds__(128)
void wmma_gemm_f16(const float* __restrict__ X, const float* __restrict__ W,
                   const float* __restrict__ bias, float* __restrict__ Y,
                   int relu)
{
    __shared__ __align__(16) _Float16 lA[16 * K];     // row-major X tile (f16)
    __shared__ __align__(16) _Float16 lBt[64 * K];    // W transposed: [col][k] (f16)

    const int tid = threadIdx.x;

    // ---- stage weights ----
#if USE_TDM_BUILTIN
    if constexpr (K <= 128) {
        __shared__ __align__(16) float lW[K * 64];    // TDM lands f32 tile here
        if (tid < 32) {
            // Build D# for a 2D tensor: dim0=64 (row), dim1=K, stride0=64,
            // tile = whole matrix, data_size=4B, count=1, type=2 ("image").
            const unsigned long long ga = (unsigned long long)W;
            const unsigned ldsOff = (unsigned)(unsigned long long)(void*)lW;
            u32x4 g0 = { 1u,                                   // count=1
                         ldsOff,                               // lds_addr [63:32]
                         (unsigned)(ga & 0xFFFFFFFFu),         // global_addr lo
                         (unsigned)((ga >> 32) & 0x1FFFFFFu) | (2u << 30) };
            i32x8 g1 = { (int)(2u << 16),                      // data_size=4B, mask=0
                         (int)(64u << 16),                     // tensor_dim0=64 -> bits 63:48
                         (int)(((unsigned)K & 0xFFFFu) << 16), // tensor_dim1 lo -> bits 95:80
                         (int)((64u << 16) | (((unsigned)K >> 16) & 0xFFFFu)), // tile_dim0=64 | dim1 hi
                         (int)K,                               // tile_dim1=K, tile_dim2=0
                         64,                                   // tensor_dim0_stride lo
                         0, 0 };                               // stride hi / dim1_stride
            i32x4 gz = { 0, 0, 0, 0 };
            __builtin_amdgcn_tensor_load_to_lds(g0, g1, gz, gz, 0);
            __builtin_amdgcn_s_wait_tensorcnt(0);
        }
        __syncthreads();
        for (int i = tid; i < 64 * K; i += 128) {
            const int k = i >> 6, c = i & 63;
            lBt[c * K + k] = (_Float16)lW[k * 64 + c];
        }
    } else {
        for (int i = tid; i < 64 * K; i += 128) {
            const int k = i >> 6, c = i & 63;
            lBt[c * K + k] = (_Float16)W[(size_t)k * 64 + c];
        }
    }
#else
    for (int i = tid; i < 64 * K; i += 128) {
        const int k = i >> 6, c = i & 63;
        lBt[c * K + k] = (_Float16)W[(size_t)k * 64 + c];
    }
#endif

    // ---- stage X tile (16 rows x K), f32 -> f16, coalesced ----
    const int rowBase = blockIdx.x * 16;
    for (int i = tid; i < 16 * K; i += 128) {
        const int r = i / K, k = i % K;
        lA[r * K + k] = (_Float16)X[(size_t)(rowBase + r) * K + k];
    }
    __syncthreads();

    const int lane = tid & 31;
    const int wave = tid >> 5;          // column tile
    const int half = lane >> 4;
    const int l16  = lane & 15;
    const int colBase = wave * 16;

    v8f acc = {};
#pragma unroll
    for (int k0 = 0; k0 < K; k0 += 32) {
        // A fragment: lanes 0-15 K=0..7,16..23 ; lanes 16-31 K=8..15,24..31
        const _Float16* ap = lA + l16 * K + k0 + half * 8;
        v8h alo = *(const v8h*)(ap);
        v8h ahi = *(const v8h*)(ap + 16);
        v16h a = __builtin_shufflevector(alo, ahi, 0,1,2,3,4,5,6,7,8,9,10,11,12,13,14,15);
        // B fragment: lanes 0-15 K=0..7,16..23 ; lanes 16-31 K=8..15,24..31, col = colBase+l16
        const _Float16* bp = lBt + (colBase + l16) * K + k0 + half * 8;
        v8h blo = *(const v8h*)(bp);
        v8h bhi = *(const v8h*)(bp + 16);
        v16h b = __builtin_shufflevector(blo, bhi, 0,1,2,3,4,5,6,7,8,9,10,11,12,13,14,15);

        acc = __builtin_amdgcn_wmma_f32_16x16x32_f16(
            false, a, false, b, (short)0, acc, false, false);
    }

    const float bb = bias ? bias[colBase + l16] : 0.0f;
#pragma unroll
    for (int r = 0; r < 8; ++r) {
        float v = acc[r] + bb;
        if (relu) v = fmaxf(v, 0.0f);
        Y[(size_t)(rowBase + r + 8 * half) * HID + (colBase + l16)] = v;
    }
}

// ---------------------------------------------------------------------------
// Order-preserving float <-> uint encoding for atomic max on scores.
// ---------------------------------------------------------------------------
__device__ __forceinline__ unsigned enc_f32(float f) {
    unsigned u = __float_as_uint(f);
    return (u & 0x80000000u) ? ~u : (u | 0x80000000u);
}
__device__ __forceinline__ float dec_f32(unsigned u) {
    return __uint_as_float((u & 0x80000000u) ? (u & 0x7FFFFFFFu) : ~u);
}

// score[e][h] = (q[dst] . k[src]) * 0.25 per head; atomicMax into mEnc[dst][h]
__global__ void edge_score_max(const float* __restrict__ q, const float* __restrict__ k,
                               const int* __restrict__ src, const int* __restrict__ dst,
                               float* __restrict__ score, unsigned* __restrict__ mEnc)
{
    int e = blockIdx.x * blockDim.x + threadIdx.x;
    if (e >= EE) return;
    const int s = src[e], d = dst[e];
    const float4* qd = (const float4*)(q + (size_t)d * HID);
    const float4* ks = (const float4*)(k + (size_t)s * HID);
#pragma unroll
    for (int h = 0; h < 4; ++h) {
        float acc = 0.0f;
#pragma unroll
        for (int j = 0; j < 4; ++j) {
            float4 a = qd[h * 4 + j], b = ks[h * 4 + j];
            acc += a.x * b.x + a.y * b.y + a.z * b.z + a.w * b.w;
        }
        acc *= 0.25f;   // 1/sqrt(D_HEAD=16)
        score[(size_t)e * 4 + h] = acc;
        atomicMax(&mEnc[(size_t)d * 4 + h], enc_f32(acc));
    }
}

// e = exp(score - m[dst]) in place; atomicAdd into denom[dst][h]
__global__ void edge_exp_sum(float* __restrict__ score, const unsigned* __restrict__ mEnc,
                             const int* __restrict__ dst, float* __restrict__ denom)
{
    int e = blockIdx.x * blockDim.x + threadIdx.x;
    if (e >= EE) return;
    const int d = dst[e];
#pragma unroll
    for (int h = 0; h < 4; ++h) {
        float m  = dec_f32(mEnc[(size_t)d * 4 + h]);
        float ex = __expf(score[(size_t)e * 4 + h] - m);
        score[(size_t)e * 4 + h] = ex;
        atomicAdd(&denom[(size_t)d * 4 + h], ex);
    }
}

// agg[dst][:] += (e/denom[dst]) * v[src][:]   (one 64-thread block per edge)
__global__ __launch_bounds__(64)
void edge_aggregate(const float* __restrict__ escore, const float* __restrict__ denom,
                    const float* __restrict__ v, const int* __restrict__ src,
                    const int* __restrict__ dst, float* __restrict__ agg)
{
    const int e = blockIdx.x;
    const int t = threadIdx.x;          // 0..63, head = t>>4
    const int s = src[e], d = dst[e];
    const int h = t >> 4;
    const float a = escore[(size_t)e * 4 + h] / denom[(size_t)d * 4 + h];
    atomicAdd(&agg[(size_t)d * HID + t], a * v[(size_t)s * HID + t]);
}

// h = layernorm(relu(agg + skip)) * gamma + beta   (one 64-thread block/node)
__global__ __launch_bounds__(64)
void finish_layer(const float* __restrict__ agg, const float* __restrict__ skip,
                  const float* __restrict__ gamma, const float* __restrict__ beta,
                  float* __restrict__ h)
{
    const int n = blockIdx.x;
    const int t = threadIdx.x;
    __shared__ float red[64];

    float val = fmaxf(agg[(size_t)n * HID + t] + skip[(size_t)n * HID + t], 0.0f);

    red[t] = val;
    __syncthreads();
    for (int o = 32; o > 0; o >>= 1) { if (t < o) red[t] += red[t + o]; __syncthreads(); }
    const float mu = red[0] * (1.0f / 64.0f);
    __syncthreads();

    const float dv = val - mu;
    red[t] = dv * dv;
    __syncthreads();
    for (int o = 32; o > 0; o >>= 1) { if (t < o) red[t] += red[t + o]; __syncthreads(); }
    const float inv = rsqrtf(red[0] * (1.0f / 64.0f) + 1e-5f);

    h[(size_t)n * HID + t] = dv * inv * gamma[t] + beta[t];
}

// cat[n][0:64] = h[n], cat[n][64:128] = hg[n]
__global__ void concat_kernel(const float* __restrict__ h, const float* __restrict__ hg,
                              float* __restrict__ cat)
{
    size_t i = (size_t)blockIdx.x * blockDim.x + threadIdx.x;
    if (i >= (size_t)NN * 128) return;
    size_t n = i >> 7;
    int j = (int)(i & 127);
    cat[i] = (j < 64) ? h[n * 64 + j] : hg[n * 64 + (j - 64)];
}

// Four tiny classifier heads (64 -> 3,2,2,2), one thread per node.
__global__ __launch_bounds__(256)
void heads_kernel(const float* __restrict__ hf,
                  const float* __restrict__ Wd, const float* __restrict__ bd,
                  const float* __restrict__ Wi, const float* __restrict__ bi,
                  const float* __restrict__ Ws, const float* __restrict__ bs,
                  const float* __restrict__ Wm, const float* __restrict__ bm,
                  float* __restrict__ out)
{
    int n = blockIdx.x * blockDim.x + threadIdx.x;
    if (n >= NN) return;
    float x[64];
    const float4* xp = (const float4*)(hf + (size_t)n * HID);
#pragma unroll
    for (int j = 0; j < 16; ++j) {
        float4 t = xp[j];
        x[4 * j] = t.x; x[4 * j + 1] = t.y; x[4 * j + 2] = t.z; x[4 * j + 3] = t.w;
    }
    float* o0 = out;
    float* o1 = out + (size_t)NN * 3;
    float* o2 = o1 + (size_t)NN * 2;
    float* o3 = o2 + (size_t)NN * 2;
#pragma unroll
    for (int c = 0; c < 3; ++c) {
        float acc = bd[c];
#pragma unroll
        for (int j = 0; j < 64; ++j) acc += x[j] * Wd[j * 3 + c];
        o0[(size_t)n * 3 + c] = acc;
    }
#pragma unroll
    for (int c = 0; c < 2; ++c) {
        float a1 = bi[c], a2 = bs[c], a3 = bm[c];
#pragma unroll
        for (int j = 0; j < 64; ++j) {
            a1 += x[j] * Wi[j * 2 + c];
            a2 += x[j] * Ws[j * 2 + c];
            a3 += x[j] * Wm[j * 2 + c];
        }
        o1[(size_t)n * 2 + c] = a1;
        o2[(size_t)n * 2 + c] = a2;
        o3[(size_t)n * 2 + c] = a3;
    }
}

// ---------------------------------------------------------------------------
extern "C" void kernel_launch(void* const* d_in, const int* in_sizes, int n_in,
                              void* d_out, int out_size, void* d_ws, size_t ws_size,
                              hipStream_t stream)
{
    (void)in_sizes; (void)n_in; (void)out_size; (void)ws_size;

    const float* x    = (const float*)d_in[0];   // N x 256
    const float* gf   = (const float*)d_in[1];   // N x 64
    const int*   esrc = (const int*)d_in[2];     // E
    const int*   edst = (const int*)d_in[3];     // E
    const float* Win  = (const float*)d_in[4];   // 256 x 64
    const float* bin  = (const float*)d_in[5];
    const float* Wg   = (const float*)d_in[6];   // 64 x 64
    const float* bg   = (const float*)d_in[7];
    const float* Wf   = (const float*)d_in[28];  // 128 x 64
    const float* bf   = (const float*)d_in[29];
    const float* Wd   = (const float*)d_in[30];  const float* bd = (const float*)d_in[31];
    const float* Wi   = (const float*)d_in[32];  const float* bi = (const float*)d_in[33];
    const float* Wsec = (const float*)d_in[34];  const float* bsec = (const float*)d_in[35];
    const float* Wm   = (const float*)d_in[36];  const float* bm = (const float*)d_in[37];

    float* ws = (float*)d_ws;
    const size_t NF = (size_t)NN * HID;
    float*    h      = ws;                                  // N x 64
    float*    hg     = h    + NF;                           // N x 64
    float*    q      = hg   + NF;                           // N x 64
    float*    kbuf   = q    + NF;                           // N x 64
    float*    vbuf   = kbuf + NF;                           // N x 64
    float*    skip   = vbuf + NF;                           // N x 64
    float*    agg    = skip + NF;                           // N x 64
    float*    escore = agg  + NF;                           // E x 4
    unsigned* mEnc   = (unsigned*)(escore + (size_t)EE * 4);// N x 4
    float*    denom  = (float*)(mEnc + (size_t)NN * 4);     // N x 4
    float*    cat    = q;     // reuse q+k region: N x 128 contiguous
    float*    hf     = vbuf;  // reuse v region:   N x 64

    const dim3 gGemm(NN / 16), bGemm(128);

    wmma_gemm_f16<256><<<gGemm, bGemm, 0, stream>>>(x,  Win, bin, h,  0);
    wmma_gemm_f16<64> <<<gGemm, bGemm, 0, stream>>>(gf, Wg,  bg,  hg, 0);

    for (int layer = 0; layer < 2; ++layer) {
        const float* Wq  = (const float*)d_in[8 + layer * 8 + 0];
        const float* bq  = (const float*)d_in[8 + layer * 8 + 1];
        const float* Wk  = (const float*)d_in[8 + layer * 8 + 2];
        const float* bk  = (const float*)d_in[8 + layer * 8 + 3];
        const float* Wv  = (const float*)d_in[8 + layer * 8 + 4];
        const float* bv  = (const float*)d_in[8 + layer * 8 + 5];
        const float* Wsk = (const float*)d_in[8 + layer * 8 + 6];
        const float* bsk = (const float*)d_in[8 + layer * 8 + 7];
        const float* gamma = (const float*)d_in[24 + layer * 2 + 0];
        const float* beta  = (const float*)d_in[24 + layer * 2 + 1];

        wmma_gemm_f16<64><<<gGemm, bGemm, 0, stream>>>(h, Wq,  bq,  q,    0);
        wmma_gemm_f16<64><<<gGemm, bGemm, 0, stream>>>(h, Wk,  bk,  kbuf, 0);
        wmma_gemm_f16<64><<<gGemm, bGemm, 0, stream>>>(h, Wv,  bv,  vbuf, 0);
        wmma_gemm_f16<64><<<gGemm, bGemm, 0, stream>>>(h, Wsk, bsk, skip, 0);

        hipMemsetAsync(mEnc,  0, (size_t)NN * 4 * sizeof(unsigned), stream);
        hipMemsetAsync(denom, 0, (size_t)NN * 4 * sizeof(float), stream);
        hipMemsetAsync(agg,   0, NF * sizeof(float), stream);

        edge_score_max<<<(EE + 255) / 256, 256, 0, stream>>>(q, kbuf, esrc, edst, escore, mEnc);
        edge_exp_sum  <<<(EE + 255) / 256, 256, 0, stream>>>(escore, mEnc, edst, denom);
        edge_aggregate<<<EE, 64, 0, stream>>>(escore, denom, vbuf, esrc, edst, agg);
        finish_layer  <<<NN, 64, 0, stream>>>(agg, skip, gamma, beta, h);
    }

    concat_kernel<<<(int)(((size_t)NN * 128 + 255) / 256), 256, 0, stream>>>(h, hg, cat);
    wmma_gemm_f16<128><<<gGemm, bGemm, 0, stream>>>(cat, Wf, bf, hf, 1);
    heads_kernel<<<(NN + 255) / 256, 256, 0, stream>>>(hf, Wd, bd, Wi, bi, Wsec, bsec,
                                                       Wm, bm, (float*)d_out);
}
